// MultiHeadSelfAttention_33200097198632
// MI455X (gfx1250) — matrix operand order repaired and checked by reading, not compiled, not run on hardware
//
#include <hip/hip_runtime.h>
#include <hip/hip_bf16.h>

// MHSA forward for MI455X (gfx1250, wave32, WMMA).
// - All four matmuls on v_wmma_f32_16x16x32_bf16 (f32 accumulation).
// - Flash attention in the transposed formulation (C-layout cols = query =
//   lane%16) so softmax stats are per-lane scalars; only cross-lane op is a
//   ds_swizzle SWAPX16.
// - K/V tiles staged into LDS once per block (shared by 4 waves) with
//   GLOBAL_LOAD_ASYNC_TO_LDS_B128 (ASYNCcnt), double-buffered so the DMA for
//   step i+1 overlaps compute of step i.

#define B_ 4
#define S_ 2048
#define E_ 1024
#define H_ 16
#define D_ 64

typedef __bf16 bf16_t;
typedef __attribute__((ext_vector_type(16))) __bf16 v16bf;
typedef __attribute__((ext_vector_type(8)))  __bf16 v8bf;
typedef __attribute__((ext_vector_type(8)))  float  v8f;

static __device__ __forceinline__ v8f wmma_bf16(v16bf a, v16bf b, v8f c) {
  // 8 args: (neg_a, A, neg_b, B, c_mod, C, reuse_a, reuse_b)
  return __builtin_amdgcn_wmma_f32_16x16x32_bf16(false, a, false, b,
                                                 (short)0, c, false, false);
}

// Exchange with lane^16 partner (SWAPX16: xor=0x10, and=0x1f -> 0x401F).
static __device__ __forceinline__ float swap16(float v) {
  return __int_as_float(__builtin_amdgcn_ds_swizzle(__float_as_int(v), 0x401F));
}

static __device__ __forceinline__ v16bf combine8(v8bf lo, v8bf hi) {
  v16bf r;
#pragma unroll
  for (int j = 0; j < 8; ++j) { r[j] = lo[j]; r[8 + j] = hi[j]; }
  return r;
}

// Async DMA: 16 bytes per lane, global -> LDS.  Tracked by ASYNCcnt.
// VDST = VGPR holding the LDS byte address (low 32 bits of the generic
// pointer per the ISA aperture rules), ADDR = 64-bit global address.
static __device__ __forceinline__ void async_copy16(const void* gaddr,
                                                    unsigned lds_addr) {
  asm volatile("global_load_async_to_lds_b128 %0, %1, off"
               :: "v"(lds_addr), "v"(gaddr) : "memory");
}
#define WAIT_ASYNCCNT(n) asm volatile("s_wait_asynccnt " #n ::: "memory")

// ---------------------------------------------------------------------------
// Kernel 0: weight conversion.
//   WqT/WkT/WvT[h][e][d] = bf16(W[h][d][e])   (B-frag reads contiguous in d)
//   WoB[f][e]            = bf16(Wo[f][e])     (already contiguous in K=e)
// ---------------------------------------------------------------------------
__global__ __launch_bounds__(256) void cvt_weights(
    const float* __restrict__ Wq, const float* __restrict__ Wk,
    const float* __restrict__ Wv, const float* __restrict__ Wo,
    bf16_t* __restrict__ WqT, bf16_t* __restrict__ WkT,
    bf16_t* __restrict__ WvT, bf16_t* __restrict__ WoB) {
  int tid = blockIdx.x * blockDim.x + threadIdx.x;   // grid covers E*E
  if (tid < H_ * D_ * D_) {
    int h = tid >> 12, rem = tid & 4095, d = rem >> 6, e = rem & 63;
    int dst = h * 4096 + e * 64 + d;
    WqT[dst] = (bf16_t)Wq[tid];
    WkT[dst] = (bf16_t)Wk[tid];
    WvT[dst] = (bf16_t)Wv[tid];
  }
  WoB[tid] = (bf16_t)Wo[tid];
}

// ---------------------------------------------------------------------------
// Kernel 1: per-head QKV projection.
// One wave: 16 s-rows x 64 e-cols for one (b,h); q,k,v share the A fragment.
//   qh[b,h,s,d] = (x_h @ Wq + bq) * 1/sqrt(D)   (scale folded in)
//   kh[b,h,s,d] =  x_h @ Wk + bk
//   vT[b,h,d,s] =  x_h @ Wv + bv                (stored transposed)
// ---------------------------------------------------------------------------
__global__ __launch_bounds__(128) void qkv_proj(
    const float* __restrict__ x,
    const bf16_t* __restrict__ WqT, const bf16_t* __restrict__ WkT,
    const bf16_t* __restrict__ WvT,
    const float* __restrict__ bq, const float* __restrict__ bk,
    const float* __restrict__ bv,
    bf16_t* __restrict__ qh, bf16_t* __restrict__ kh,
    bf16_t* __restrict__ vT) {
  const int lane = threadIdx.x & 31, wave = threadIdx.x >> 5;
  const int lid = lane & 15, hlf = lane >> 4;
  const int bh  = blockIdx.x >> 5;            // b*H + h
  const int grp = blockIdx.x & 31;
  const int h = bh & (H_ - 1);
  const int b = bh >> 4;
  const int s0 = (grp * 4 + wave) * 16;

  // A fragments: x rows (M=s, K=d), fp32 -> bf16.
  const float* xrow = x + ((size_t)(b * S_ + s0 + lid)) * E_ + h * D_;
  v16bf a0, a1;
#pragma unroll
  for (int kk = 0; kk < 2; ++kk) {
    const int dOff = kk * 32;
    v8f lo = *(const v8f*)(xrow + dOff + hlf * 16);
    v8f hi = *(const v8f*)(xrow + dOff + hlf * 16 + 8);
#pragma unroll
    for (int j = 0; j < 8; ++j) {
      if (kk == 0) { a0[j] = (bf16_t)lo[j]; a0[8 + j] = (bf16_t)hi[j]; }
      else         { a1[j] = (bf16_t)lo[j]; a1[8 + j] = (bf16_t)hi[j]; }
    }
  }

  const bf16_t* Wts[3]    = {WqT, WkT, WvT};
  const float*  biases[3] = {bq, bk, bv};
#pragma unroll
  for (int p = 0; p < 3; ++p) {
    const bf16_t* Wt   = Wts[p] + h * 4096;     // [e][d]
    const float*  bias = biases[p] + h * 64;
#pragma unroll
    for (int eT = 0; eT < 4; ++eT) {
      const int e = eT * 16 + lid;
      v16bf b0 = *(const v16bf*)(Wt + e * 64 + 0  + hlf * 16);
      v16bf b1 = *(const v16bf*)(Wt + e * 64 + 32 + hlf * 16);
      v8f acc = {};
      acc = wmma_bf16(a0, b0, acc);
      acc = wmma_bf16(a1, b1, acc);
      const float bval = bias[e];
      if (p == 0) {
        bf16_t* dst = qh + ((size_t)bh * S_ + s0 + hlf * 8) * D_ + e;
#pragma unroll
        for (int r = 0; r < 8; ++r)
          dst[(size_t)r * D_] = (bf16_t)((acc[r] + bval) * 0.125f);
      } else if (p == 1) {
        bf16_t* dst = kh + ((size_t)bh * S_ + s0 + hlf * 8) * D_ + e;
#pragma unroll
        for (int r = 0; r < 8; ++r)
          dst[(size_t)r * D_] = (bf16_t)(acc[r] + bval);
      } else {
        bf16_t* dst = vT + ((size_t)bh * D_ + e) * S_ + s0 + hlf * 8;
        v8bf pk;
#pragma unroll
        for (int r = 0; r < 8; ++r) pk[r] = (bf16_t)(acc[r] + bval);
        *(v8bf*)dst = pk;   // contiguous in s
      }
    }
  }
}

// ---------------------------------------------------------------------------
// Kernel 2: flash attention, transposed formulation, async-LDS staged K/V.
// Block = 4 waves, same (b,h), 64 consecutive queries (16 per wave).
// Per 32-key step: K tile (32x64) and V^T tile (64x32) are DMA'd to LDS once
// per block (double-buffered), all 4 waves read fragments via ds_load.
//   S^T tile = K(16k x 64d) @ Q^T(64d x 16q)   -> C cols = query = lane%16
//   O^T     += V^T(16d x 32k) @ P^T(32k x 16q)
// ---------------------------------------------------------------------------
__global__ __launch_bounds__(128) void attention(
    const bf16_t* __restrict__ qh, const bf16_t* __restrict__ kh,
    const bf16_t* __restrict__ vT, bf16_t* __restrict__ ctx) {
  __shared__ bf16_t Kb[2][32 * 64];   // [key][d]   4 KB per buffer
  __shared__ bf16_t Vb[2][64 * 32];   // [d][key]   4 KB per buffer

  const int tid  = threadIdx.x;
  const int lane = tid & 31, wave = tid >> 5;
  const int lid = lane & 15, hlf = lane >> 4;
  const int bh = blockIdx.x >> 5;
  const int qg = blockIdx.x & 31;
  const int q0 = (qg * 4 + wave) * 16;
  const int b = bh >> 4, h = bh & 15;

  const bf16_t* kbase = kh + (size_t)bh * S_ * D_;
  const bf16_t* vbase = vT + (size_t)bh * D_ * S_;

  // Q^T B fragments (col N = query, K = d sequential per lane-half).
  const bf16_t* qrow = qh + ((size_t)bh * S_ + q0 + lid) * D_;
  const v16bf qb0 = *(const v16bf*)(qrow + 0  + hlf * 16);
  const v16bf qb1 = *(const v16bf*)(qrow + 32 + hlf * 16);

  float mrow = -3.0e38f, lrow = 0.0f;
  v8f o[4];
#pragma unroll
  for (int n = 0; n < 4; ++n)
#pragma unroll
    for (int r = 0; r < 8; ++r) o[n][r] = 0.0f;

  // ---- cooperative async stage of one 32-key tile into buffer `buf` ----
  auto stage = [&](int buf, int kv) {
    // K tile: rows kv..kv+31, all 64 d -> contiguous 4 KB block.
    const bf16_t* gk = kbase + (size_t)kv * D_;
    unsigned kl = (unsigned)(size_t)&Kb[buf][0];
    async_copy16(gk + tid * 8,         kl + tid * 16);
    async_copy16(gk + (128 + tid) * 8, kl + (128 + tid) * 16);
    // V^T tile: 64 rows (d) x 32 keys, row stride S_ in memory.
    unsigned vl = (unsigned)(size_t)&Vb[buf][0];
#pragma unroll
    for (int half = 0; half < 2; ++half) {
      int c = tid + half * 128;             // 256 chunks of 16B
      int row = c >> 2, col8 = (c & 3) * 8;
      async_copy16(vbase + (size_t)row * S_ + kv + col8, vl + c * 16);
    }
  };

  stage(0, 0);
  const int NSTEP = S_ / 32;
  for (int i = 0; i < NSTEP; ++i) {
    const int cur = i & 1;
    if (i + 1 < NSTEP) {
      stage(cur ^ 1, (i + 1) * 32);
      WAIT_ASYNCCNT(4);   // per-wave in-order: first 4 (current buf) done
    } else {
      WAIT_ASYNCCNT(0);
    }
    __syncthreads();      // current buffer visible to all waves

    // ---- S^T = K @ Q^T : two 16x16 tiles from LDS ----
    v8f sA = {}, sB = {};
    const bf16_t* kr0 = &Kb[cur][(size_t)lid * 64];
    const bf16_t* kr1 = &Kb[cur][(size_t)(16 + lid) * 64];
#pragma unroll
    for (int kk = 0; kk < 2; ++kk) {
      const int dOff = kk * 32;
      v16bf ka = combine8(*(const v8bf*)(kr0 + dOff + hlf * 16),
                          *(const v8bf*)(kr0 + dOff + hlf * 16 + 8));
      v16bf kb = combine8(*(const v8bf*)(kr1 + dOff + hlf * 16),
                          *(const v8bf*)(kr1 + dOff + hlf * 16 + 8));
      sA = wmma_bf16(ka, kk ? qb1 : qb0, sA);
      sB = wmma_bf16(kb, kk ? qb1 : qb0, sB);
    }

    // ---- online softmax (scale folded into q) ----
    float lm = sA[0];
#pragma unroll
    for (int r = 1; r < 8; ++r) lm = fmaxf(lm, sA[r]);
#pragma unroll
    for (int r = 0; r < 8; ++r) lm = fmaxf(lm, sB[r]);
    lm = fmaxf(lm, swap16(lm));
    const float mn   = fmaxf(mrow, lm);
    const float corr = __expf(mrow - mn);
    float p0[8], p1[8], ls = 0.0f;
#pragma unroll
    for (int r = 0; r < 8; ++r) {
      p0[r] = __expf(sA[r] - mn);
      p1[r] = __expf(sB[r] - mn);
      ls += p0[r] + p1[r];
    }
    ls += swap16(ls);
    lrow = lrow * corr + ls;
    mrow = mn;
#pragma unroll
    for (int n = 0; n < 4; ++n)
#pragma unroll
      for (int r = 0; r < 8; ++r) o[n][r] *= corr;

    // ---- P^T B fragment (col = query, K = key sequential per half) ----
    v16bf pt;
#pragma unroll
    for (int r = 0; r < 8; ++r) {
      const float x0 = swap16(p0[r]);
      const float x1 = swap16(p1[r]);
      pt[r]     = (bf16_t)(hlf ? x1    : p0[r]);
      pt[8 + r] = (bf16_t)(hlf ? p1[r] : x0);
    }

    // ---- O^T += V^T @ P^T : 4 d-tiles of 16, V^T from LDS ----
#pragma unroll
    for (int dblk = 0; dblk < 4; ++dblk) {
      const bf16_t* vr = &Vb[cur][(size_t)(dblk * 16 + lid) * 32];
      v16bf va = combine8(*(const v8bf*)(vr + hlf * 16),
                          *(const v8bf*)(vr + hlf * 16 + 8));
      o[dblk] = wmma_bf16(va, pt, o[dblk]);
    }

    __syncthreads();      // all waves done reading before buffer re-staged
  }

  // ---- normalize and store ctx[b,s=q,h*64+d] ----
  const float inv = 1.0f / lrow;
  bf16_t* crow = ctx + ((size_t)(b * S_ + q0 + lid)) * E_ + h * D_ + hlf * 8;
#pragma unroll
  for (int dblk = 0; dblk < 4; ++dblk) {
    v8bf pk;
#pragma unroll
    for (int r = 0; r < 8; ++r) pk[r] = (bf16_t)(o[dblk][r] * inv);
    *(v8bf*)(crow + dblk * 16) = pk;
  }
}

// ---------------------------------------------------------------------------
// Kernel 3: output projection  out[m,f] = ctx[m,:] @ Wo[f,:]^T + bo[f]
// Per wave: 16 rows x 64 cols, K = 1024 -> 128 WMMAs.
// ---------------------------------------------------------------------------
__global__ __launch_bounds__(128) void out_proj(
    const bf16_t* __restrict__ ctx, const bf16_t* __restrict__ WoB,
    const float* __restrict__ bo, float* __restrict__ out) {
  const int lane = threadIdx.x & 31, wave = threadIdx.x >> 5;
  const int lid = lane & 15, hlf = lane >> 4;
  const int rt = blockIdx.x >> 2;
  const int cq = blockIdx.x & 3;
  const int colBase = (cq * 4 + wave) * 64;

  v8f acc[4];
#pragma unroll
  for (int n = 0; n < 4; ++n) {
    const float bval = bo[colBase + n * 16 + lid];
#pragma unroll
    for (int r = 0; r < 8; ++r) acc[n][r] = bval;
  }

  const bf16_t* arow = ctx + ((size_t)rt * 16 + lid) * E_;
  for (int e0 = 0; e0 < E_; e0 += 32) {
    v16bf va = combine8(*(const v8bf*)(arow + e0 + hlf * 16),
                        *(const v8bf*)(arow + e0 + hlf * 16 + 8));
#pragma unroll
    for (int n = 0; n < 4; ++n) {
      v16bf bw = *(const v16bf*)(WoB + (size_t)(colBase + n * 16 + lid) * E_ +
                                 e0 + hlf * 16);
      acc[n] = wmma_bf16(va, bw, acc[n]);
    }
  }

#pragma unroll
  for (int n = 0; n < 4; ++n) {
    float* dst = out + ((size_t)rt * 16 + hlf * 8) * E_ + colBase + n * 16 + lid;
#pragma unroll
    for (int r = 0; r < 8; ++r) dst[(size_t)r * E_] = acc[n][r];
  }
}

// ---------------------------------------------------------------------------
extern "C" void kernel_launch(void* const* d_in, const int* in_sizes, int n_in,
                              void* d_out, int out_size, void* d_ws,
                              size_t ws_size, hipStream_t stream) {
  (void)in_sizes; (void)n_in; (void)out_size; (void)ws_size;
  const float* x  = (const float*)d_in[0];
  const float* Wq = (const float*)d_in[1];
  const float* bq = (const float*)d_in[2];
  const float* Wk = (const float*)d_in[3];
  const float* bk = (const float*)d_in[4];
  const float* Wv = (const float*)d_in[5];
  const float* bv = (const float*)d_in[6];
  const float* Wo = (const float*)d_in[7];
  const float* bo = (const float*)d_in[8];

  char* ws = (char*)d_ws;
  const size_t MB = 1024 * 1024;
  bf16_t* qh  = (bf16_t*)(ws);                       // [B,H,S,D]  16 MB
  bf16_t* kh  = (bf16_t*)(ws + 16 * MB);             // [B,H,S,D]  16 MB
  bf16_t* vT  = (bf16_t*)(ws + 32 * MB);             // [B,H,D,S]  16 MB
  bf16_t* ctx = (bf16_t*)(ws + 48 * MB);             // [B,S,E]    16 MB
  bf16_t* WqT = (bf16_t*)(ws + 64 * MB);             // [H,D,D]   128 KB
  bf16_t* WkT = (bf16_t*)(ws + 64 * MB + 128 * 1024);
  bf16_t* WvT = (bf16_t*)(ws + 64 * MB + 256 * 1024);
  bf16_t* WoB = (bf16_t*)(ws + 64 * MB + 384 * 1024);// [E,E]       2 MB

  cvt_weights<<<(E_ * E_) / 256, 256, 0, stream>>>(Wq, Wk, Wv, Wo,
                                                   WqT, WkT, WvT, WoB);
  qkv_proj<<<B_ * H_ * (S_ / 64), 128, 0, stream>>>(x, WqT, WkT, WvT,
                                                    bq, bk, bv, qh, kh, vT);
  attention<<<B_ * H_ * (S_ / 64), 128, 0, stream>>>(qh, kh, vT, ctx);
  out_proj<<<(B_ * S_ / 16) * 4, 128, 0, stream>>>(ctx, WoB, bo,
                                                   (float*)d_out);
}
